// GraphNode_26405458936406
// MI455X (gfx1250) — matrix operand order repaired and checked by reading, not compiled, not run on hardware
//
#include <hip/hip_runtime.h>
#include <hip/hip_bf16.h>

#define N_NODES  50000
#define N_EDGES  800000
#define NODE_DIM 128
#define EDGE_DIM 32
#define GNN_DIM  128
#define K1 (NODE_DIM + EDGE_DIM)   // 160
#define K2 (NODE_DIM + GNN_DIM)    // 256
#define LDA1 168                   // padded LDS row stride (halfs), edge kernel
#define LDA2 264                   // padded LDS row stride (halfs), node kernel

typedef __attribute__((ext_vector_type(16))) __bf16        v16bf;
typedef __attribute__((ext_vector_type(8)))  float         v8f;
typedef __attribute__((ext_vector_type(4)))  unsigned int  u32x4;

union FragBF { v16bf v; u32x4 q[2]; };

// round-to-nearest-even f32 -> bf16 bits
static __device__ inline unsigned short f2bf(float f) {
    unsigned int u = __builtin_bit_cast(unsigned int, f);
    u += 0x7FFFu + ((u >> 16) & 1u);
    return (unsigned short)(u >> 16);
}

// ---------------------------------------------------------------------------
// Prep: transpose + convert W1 (160x128) -> W1T bf16 [128][160],
//                       W2 (256x128) -> W2T bf16 [128][256]
// ---------------------------------------------------------------------------
__global__ void prep_weights(const float* __restrict__ W1,
                             const float* __restrict__ W2,
                             unsigned short* __restrict__ w1t,
                             unsigned short* __restrict__ w2t) {
    int i = blockIdx.x * blockDim.x + threadIdx.x;
    if (i < GNN_DIM * K1) {
        int n = i / K1, k = i - n * K1;
        w1t[i] = f2bf(W1[k * GNN_DIM + n]);
    }
    int j = i - GNN_DIM * K1;
    if (j >= 0 && j < GNN_DIM * K2) {
        int n = j / K2, k = j - n * K2;
        w2t[j] = f2bf(W2[k * GNN_DIM + n]);
    }
}

// ---------------------------------------------------------------------------
// Edge kernel: msg = relu([x[row] | edge_attr] @ W1 + b1), fused scatter-add
// Block = 256 thr = 8 waves; each wave owns 16 output cols; 64 edges / tile.
// ---------------------------------------------------------------------------
__global__ __launch_bounds__(256) void edge_kernel(
    const float* __restrict__ x, const int* __restrict__ eid,
    const float* __restrict__ ea, const unsigned short* __restrict__ w1t,
    const float* __restrict__ b1, float* __restrict__ sums,
    float* __restrict__ cnt) {

    __shared__ unsigned short As[64 * LDA1];  // 21504 B (padded vs bank conflicts)
    __shared__ int rowS[64];                  // row * NODE_DIM (element offset)
    __shared__ int colS[64];                  // col * GNN_DIM  (element offset)

    const int tid  = threadIdx.x;
    const int wave = tid >> 5;
    const int lane = tid & 31;
    const int l16  = lane & 15;
    const int half = lane >> 4;
    const int ncol = wave * 16 + l16;        // this lane's output column

    // B fragments for this wave's 16 columns: K=160 -> 5 chunks of 32
    FragBF B[5];
    {
        const unsigned short* wp = w1t + ncol * K1 + half * 8;
        #pragma unroll
        for (int kk = 0; kk < 5; ++kk) {
            const u32x4* p = (const u32x4*)(wp + kk * 32);
            B[kk].q[0] = p[0];
            B[kk].q[1] = p[2];
        }
    }
    const float bias = b1[ncol];

    for (int tile = blockIdx.x; tile < N_EDGES / 64; tile += gridDim.x) {
        const int e0 = tile * 64;
        __syncthreads();  // previous iteration's reads done before overwrite
        if (tid < 64) {
            int r = eid[e0 + tid];
            int c = eid[N_EDGES + e0 + tid];
            atomicAdd(cnt + c, 1.0f);        // degree count
            rowS[tid] = r * NODE_DIM;
            colS[tid] = c * GNN_DIM;
        }
        __syncthreads();
        // stage A tile: gather x[row] (bf16) and edge_attr
        for (int i = tid; i < 64 * K1; i += 256) {
            int e = i / K1, k = i - e * K1;
            float v = (k < NODE_DIM)
                    ? x[(unsigned)(rowS[e] + k)]
                    : ea[(unsigned)((e0 + e) * EDGE_DIM + (k - NODE_DIM))];
            As[e * LDA1 + k] = f2bf(v);
        }
        __syncthreads();

        #pragma unroll
        for (int mt = 0; mt < 4; ++mt) {
            // seed accumulator with bias (broadcast along rows)
            v8f c = {bias, bias, bias, bias, bias, bias, bias, bias};
            const unsigned short* ap = &As[(mt * 16 + l16) * LDA1 + half * 8];
            #pragma unroll
            for (int kk = 0; kk < 5; ++kk) {
                FragBF a;
                a.q[0] = *(const u32x4*)(ap + kk * 32);
                a.q[1] = *(const u32x4*)(ap + kk * 32 + 16);
                c = __builtin_amdgcn_wmma_f32_16x16x32_bf16(
                        false, a.v, false, B[kk].v, (short)0, c, false, false);
            }
            // relu + fused scatter-add (sums is L2-resident: 25.6 MB)
            #pragma unroll
            for (int j = 0; j < 8; ++j) {
                int   m = half * 8 + j;
                float v = c[j] > 0.f ? c[j] : 0.f;
                unsigned off = (unsigned)(colS[mt * 16 + m] + ncol);
                atomicAdd(sums + off, v);    // saddr-form 32-bit offset atomic
            }
        }
    }
}

// ---------------------------------------------------------------------------
// Node kernel: out = relu([x | sums/max(cnt,1)] @ W2 + b2)
// ---------------------------------------------------------------------------
__global__ __launch_bounds__(256) void node_kernel(
    const float* __restrict__ x, const unsigned short* __restrict__ w2t,
    const float* __restrict__ b2, const float* __restrict__ sums,
    const float* __restrict__ cnt, float* __restrict__ out) {

    __shared__ unsigned short As[64 * LDA2];  // 33792 B (padded)
    __shared__ float invS[64];

    const int tid  = threadIdx.x;
    const int wave = tid >> 5;
    const int lane = tid & 31;
    const int l16  = lane & 15;
    const int half = lane >> 4;
    const int ncol = wave * 16 + l16;
    const int n0   = blockIdx.x * 64;

    if (tid < 64) {
        int n = n0 + tid;
        float c = (n < N_NODES) ? cnt[n] : 1.0f;
        invS[tid] = 1.0f / fmaxf(c, 1.0f);
    }
    __syncthreads();
    for (int i = tid; i < 64 * K2; i += 256) {
        int e = i >> 8, k = i & 255;
        int n = n0 + e;
        float v = 0.f;
        if (n < N_NODES) {
            v = (k < NODE_DIM)
              ? x[(unsigned)(n * NODE_DIM + k)]
              : sums[(unsigned)(n * GNN_DIM + (k - NODE_DIM))] * invS[e];
        }
        As[e * LDA2 + k] = f2bf(v);
    }
    __syncthreads();

    FragBF B[8];
    {
        const unsigned short* wp = w2t + ncol * K2 + half * 8;
        #pragma unroll
        for (int kk = 0; kk < 8; ++kk) {
            const u32x4* p = (const u32x4*)(wp + kk * 32);
            B[kk].q[0] = p[0];
            B[kk].q[1] = p[2];
        }
    }
    const float bias = b2[ncol];

    #pragma unroll
    for (int mt = 0; mt < 4; ++mt) {
        v8f c = {bias, bias, bias, bias, bias, bias, bias, bias};
        const unsigned short* ap = &As[(mt * 16 + l16) * LDA2 + half * 8];
        #pragma unroll
        for (int kk = 0; kk < 8; ++kk) {
            FragBF a;
            a.q[0] = *(const u32x4*)(ap + kk * 32);
            a.q[1] = *(const u32x4*)(ap + kk * 32 + 16);
            c = __builtin_amdgcn_wmma_f32_16x16x32_bf16(
                    false, a.v, false, B[kk].v, (short)0, c, false, false);
        }
        #pragma unroll
        for (int j = 0; j < 8; ++j) {
            int   m    = half * 8 + j;
            int   node = n0 + mt * 16 + m;
            float v    = c[j] > 0.f ? c[j] : 0.f;
            if (node < N_NODES)
                out[(unsigned)(node * GNN_DIM + ncol)] = v;
        }
    }
}

// ---------------------------------------------------------------------------
// Workspace layout (bytes):
//   [0, 25,600,000)            sums  f32 [N_NODES][GNN_DIM]
//   [25,600,000, 25,800,000)   cnt   f32 [N_NODES]
//   [25,800,192, +40,960)      W1T   bf16 [128][160]
//   [25,841,152, +65,536)      W2T   bf16 [128][256]
// ---------------------------------------------------------------------------
extern "C" void kernel_launch(void* const* d_in, const int* in_sizes, int n_in,
                              void* d_out, int out_size, void* d_ws, size_t ws_size,
                              hipStream_t stream) {
    (void)in_sizes; (void)n_in; (void)out_size; (void)ws_size;
    const float* x  = (const float*)d_in[0];
    const int*   ei = (const int*)d_in[1];
    const float* ea = (const float*)d_in[2];
    const float* W1 = (const float*)d_in[3];
    const float* b1 = (const float*)d_in[4];
    const float* W2 = (const float*)d_in[5];
    const float* b2 = (const float*)d_in[6];

    char* ws = (char*)d_ws;
    float*          sums = (float*)ws;
    float*          cnt  = (float*)(ws + (size_t)N_NODES * GNN_DIM * 4);
    unsigned short* w1t  = (unsigned short*)(ws + 25800192u);
    unsigned short* w2t  = (unsigned short*)(ws + 25841152u);

    // zero the accumulators every call (graph replays don't re-poison)
    hipMemsetAsync(ws, 0, (size_t)N_NODES * GNN_DIM * 4 + (size_t)N_NODES * 4,
                   stream);

    int prepN = GNN_DIM * (K1 + K2);
    prep_weights<<<(prepN + 255) / 256, 256, 0, stream>>>(W1, W2, w1t, w2t);

    edge_kernel<<<2048, 256, 0, stream>>>(x, ei, ea, w1t, b1, sums, cnt);

    node_kernel<<<(N_NODES + 63) / 64, 256, 0, stream>>>(x, w2t, b2, sums, cnt,
                                                         (float*)d_out);
}
